// Qwen3NextGatedDeltaNet_89678917140580
// MI455X (gfx1250) — matrix operand order repaired and checked by reading, not compiled, not run on hardware
//
#include <hip/hip_runtime.h>
#include <hip/hip_bf16.h>

// ---------------------------------------------------------------------------
// Qwen3-Next Gated DeltaNet block for MI455X (gfx1250, wave32, WMMA).
//
//  * Large GEMMs (206 + 69 GFLOP) via bf16 WMMA, double-buffered LDS tiles,
//    one barrier per K-step, A + all-8 B fragments preloaded so the 8
//    v_wmma_f32_16x16x32_bf16 issue back-to-back behind one s_wait_dscnt.
//  * Recurrent delta-rule scan: 1 block/head, state column per thread in
//    VGPRs (fully unrolled), LDS broadcast of k_t/q_t, fused RMSNorm +
//    SiLU(z) gating -> bf16 for the final WMMA GEMM.
//  * Conv(K=4)+SiLU+L2norm fused single pass (memory bound).
// Workspace: ~460 MB.
// ---------------------------------------------------------------------------

#define H_DIM    2048
#define L_SEQ    4096
#define NK_H     16
#define NV_H     32
#define DK_D     128
#define DV_D     128
#define KEY_DIM  (NK_H * DK_D)               // 2048
#define VAL_DIM  (NV_H * DV_D)               // 4096
#define QKVZ_DIM (2 * KEY_DIM + 2 * VAL_DIM) // 12288
#define BA_DIM   (2 * NV_H)                  // 64
#define CONV_DIM (2 * KEY_DIM + VAL_DIM)     // 8192
#define K_CONV   4
#define EPS_RMS  1e-6f

typedef __attribute__((ext_vector_type(16))) __bf16       v16bf;
typedef __attribute__((ext_vector_type(8)))  float        v8f;
typedef __attribute__((ext_vector_type(8)))  unsigned int v8u;

// ----------------------------- helpers -------------------------------------

__device__ __forceinline__ unsigned short f32_to_bf16_rne(float f) {
    unsigned int u = __builtin_bit_cast(unsigned int, f);
    unsigned int r = (u + 0x7FFFu + ((u >> 16) & 1u)) >> 16;
    return (unsigned short)r;
}

__device__ __forceinline__ float silu_f(float x) {
    return x / (1.0f + __expf(-x));
}

// ------------------------- f32 -> bf16 conversion --------------------------
// 4 elements/thread, packed 2x-u32 store: runs at the HBM roofline.

__global__ void __launch_bounds__(256)
cvt_f32_bf16(const float* __restrict__ in, unsigned short* __restrict__ out, int n4) {
    int i = blockIdx.x * 256 + threadIdx.x;   // index in float4 units
    if (i < n4) {
        float4 f = ((const float4*)in)[i];
        uint2 p;
        p.x = (unsigned int)f32_to_bf16_rne(f.x) | ((unsigned int)f32_to_bf16_rne(f.y) << 16);
        p.y = (unsigned int)f32_to_bf16_rne(f.z) | ((unsigned int)f32_to_bf16_rne(f.w) << 16);
        ((uint2*)out)[i] = p;
    }
}

// ------------------------- bf16 WMMA GEMM ----------------------------------
// C[M][N] = A[M][K] * B[N][K]^T  (bf16 operands, f32 accum).
// 256 threads = 8 waves; block tile 128x128; K-step 32; double-buffered LDS;
// wave w owns rows 16w..16w+15 across all 128 cols (8 accum tiles).

#define TILE_M 128
#define TILE_N 128
#define TILE_K 32

__global__ void __launch_bounds__(256)
gemm_bf16_wmma(const unsigned short* __restrict__ A,   // [M][K] bf16
               const unsigned short* __restrict__ B,   // [N][K] bf16
               float* __restrict__ C,                  // [M][N] f32
               int M, int N, int Kd) {
    __shared__ __align__(16) unsigned short Asl[2][TILE_M][TILE_K]; // 16 KB
    __shared__ __align__(16) unsigned short Bsl[2][TILE_N][TILE_K]; // 16 KB

    const int tid  = threadIdx.x;
    const int lane = tid & 31;
    const int wave = tid >> 5;   // 0..7
    const int half = lane >> 4;  // 0/1
    const int l16  = lane & 15;

    const int m0 = blockIdx.y * TILE_M;
    const int n0 = blockIdx.x * TILE_N;

    const int r  = tid >> 2;        // staging row 0..63
    const int cq = (tid & 3) * 8;   // staging ushort col: 0,8,16,24

    v8f acc[8] = {};
    uint4 sa0, sa1, sb0, sb1;       // global->LDS staging registers

    const unsigned short* Ab = A + (size_t)(m0 + r) * Kd + cq;
    const unsigned short* Ab2 = Ab + (size_t)64 * Kd;
    const unsigned short* Bb = B + (size_t)(n0 + r) * Kd + cq;
    const unsigned short* Bb2 = Bb + (size_t)64 * Kd;

    // prologue: tile 0 -> buffer 0
    sa0 = *(const uint4*)(Ab);
    sa1 = *(const uint4*)(Ab2);
    sb0 = *(const uint4*)(Bb);
    sb1 = *(const uint4*)(Bb2);
    *(uint4*)&Asl[0][r][cq]      = sa0;
    *(uint4*)&Asl[0][r + 64][cq] = sa1;
    *(uint4*)&Bsl[0][r][cq]      = sb0;
    *(uint4*)&Bsl[0][r + 64][cq] = sb1;
    __syncthreads();

    const int nsteps = Kd / TILE_K;
    const int arow = wave * 16 + l16;

    for (int s = 0; s < nsteps; ++s) {
        const int cur = s & 1;

        // issue next tile's global loads early (latency hidden by WMMA burst)
        if (s + 1 < nsteps) {
            const int kn = (s + 1) * TILE_K;
            sa0 = *(const uint4*)(Ab + kn);
            sa1 = *(const uint4*)(Ab2 + kn);
            sb0 = *(const uint4*)(Bb + kn);
            sb1 = *(const uint4*)(Bb2 + kn);
        }
        if (s + 2 < nsteps) {  // near-cache prefetch two tiles ahead
            __builtin_prefetch(Ab + (s + 2) * TILE_K, 0, 3);
            __builtin_prefetch(Bb + (s + 2) * TILE_K, 0, 3);
        }

        // Preload A fragment + all 8 B fragments, then 8 back-to-back WMMAs.
        // A frag (16x32): lanes 0-15 = rows; vgprs 0..3 hold K pairs
        // {0..7}+8*half, vgprs 4..7 hold {16..23}+8*half.
        v8u araw;
#pragma unroll
        for (int i = 0; i < 8; ++i) {
            int kb = ((i < 4) ? 2 * i : 16 + 2 * (i - 4)) + 8 * half;
            araw[i] = *(const unsigned int*)&Asl[cur][arow][kb];
        }
        v8u braw[8];
#pragma unroll
        for (int j = 0; j < 8; ++j) {
            const int bn = j * 16 + l16;
#pragma unroll
            for (int i = 0; i < 8; ++i) {
                int kb = ((i < 4) ? 2 * i : 16 + 2 * (i - 4)) + 8 * half;
                braw[j][i] = *(const unsigned int*)&Bsl[cur][bn][kb];
            }
        }
        v16bf afrag = __builtin_bit_cast(v16bf, araw);
#pragma unroll
        for (int j = 0; j < 8; ++j) {
            v16bf bfrag = __builtin_bit_cast(v16bf, braw[j]);
            acc[j] = __builtin_amdgcn_wmma_f32_16x16x32_bf16(
                false, afrag, false, bfrag, (short)0, acc[j], false, false);
        }

        // stage next tile into the other buffer; one barrier per K-step
        if (s + 1 < nsteps) {
            *(uint4*)&Asl[cur ^ 1][r][cq]      = sa0;
            *(uint4*)&Asl[cur ^ 1][r + 64][cq] = sa1;
            *(uint4*)&Bsl[cur ^ 1][r][cq]      = sb0;
            *(uint4*)&Bsl[cur ^ 1][r + 64][cq] = sb1;
        }
        __syncthreads();
    }

    // C/D layout: vgpr i holds M = i (lanes 0-15) / i+8 (lanes 16-31).
#pragma unroll
    for (int j = 0; j < 8; ++j) {
        const int col = n0 + j * 16 + l16;
#pragma unroll
        for (int i = 0; i < 8; ++i) {
            const int row = m0 + wave * 16 + i + 8 * half;
            C[(size_t)row * N + col] = acc[j][i];
        }
    }
}

// --------------------------- small ba GEMM ---------------------------------

__global__ void __launch_bounds__(64)
ba_gemm(const float* __restrict__ hs, const float* __restrict__ w_ba,
        float* __restrict__ ba) {
    const int l = blockIdx.x;
    const int n = threadIdx.x;
    const float* x = hs + (size_t)l * H_DIM;
    const float* w = w_ba + (size_t)n * H_DIM;
    float acc = 0.f;
#pragma unroll 4
    for (int k = 0; k < H_DIM; ++k) acc = fmaf(x[k], w[k], acc);
    ba[l * BA_DIM + n] = acc;
}

// ----------------- fused causal conv(K=4) + SiLU + L2-norm ------------------
// grid = (CONV_DIM/128, L), block = 128. Chunks 0..15 q (norm+scale),
// 16..31 k (norm), 32..63 v (silu only).

__global__ void __launch_bounds__(128)
conv_silu_norm(const float* __restrict__ qkvz, const float* __restrict__ conv_w,
               float* __restrict__ mixed) {
    const int l     = blockIdx.y;
    const int chunk = blockIdx.x;
    const int w     = threadIdx.x;
    const int c     = chunk * 128 + w;

    int qc;  // conv channel -> qkvz column (per-NK-head block of 768)
    if (c < KEY_DIM) {
        qc = (c >> 7) * 768 + (c & 127);
    } else if (c < 2 * KEY_DIM) {
        int cc = c - KEY_DIM;
        qc = (cc >> 7) * 768 + 128 + (cc & 127);
    } else {
        int vi = c - 2 * KEY_DIM;
        int nv = vi >> 7;
        qc = (nv >> 1) * 768 + 256 + (nv & 1) * 128 + (vi & 127);
    }

    float acc = 0.f;
#pragma unroll
    for (int j = 0; j < K_CONV; ++j) {
        int tl = l - (K_CONV - 1) + j;
        if (tl >= 0)
            acc = fmaf(qkvz[(size_t)tl * QKVZ_DIM + qc], conv_w[c * K_CONV + j], acc);
    }
    float x = silu_f(acc);

    if (chunk < 32) {
        __shared__ float red[128];
        red[w] = x * x;
        __syncthreads();
        for (int s = 64; s > 0; s >>= 1) {
            if (w < s) red[w] += red[w + s];
            __syncthreads();
        }
        float nrm = rsqrtf(red[0] + 1e-6f);
        x *= nrm;
        if (chunk < 16) x *= 0.08838834764831845f;  // DK^-0.5
    }
    mixed[(size_t)l * CONV_DIM + c] = x;
}

// ---------------------- decay g and gate beta ------------------------------

__global__ void __launch_bounds__(32)
gb_kernel(const float* __restrict__ ba, const float* __restrict__ dt_bias,
          const float* __restrict__ a_log, float* __restrict__ g,
          float* __restrict__ beta) {
    const int l  = blockIdx.x;
    const int nv = threadIdx.x;
    const int base = l * BA_DIM + (nv >> 1) * 4 + (nv & 1);
    float b = ba[base];
    float a = ba[base + 2];
    beta[l * NV_H + nv] = 1.0f / (1.0f + __expf(-b));
    float xa = a + dt_bias[nv];
    float sp = (xa > 20.0f) ? xa : log1pf(__expf(xa));
    g[l * NV_H + nv] = -__expf(a_log[nv]) * sp;
}

// --------------------- recurrent delta-rule scan ---------------------------
// One block per value head; thread v keeps S[:, v] in VGPRs.
// Fused epilogue: RMSNorm over DV + norm_w + SiLU(z) gating -> bf16.

__global__ void __launch_bounds__(128)
scan_kernel(const float* __restrict__ mixed, const float* __restrict__ qkvz,
            const float* __restrict__ g, const float* __restrict__ beta,
            const float* __restrict__ norm_w, unsigned short* __restrict__ cg) {
    const int nv   = blockIdx.x;
    const int v    = threadIdx.x;
    const int nk   = nv >> 1;
    const int lane = v & 31;
    const int wid  = v >> 5;

    __shared__ float ksh[128];
    __shared__ float qsh[128];
    __shared__ float red[4];

    float S[128];
#pragma unroll
    for (int j = 0; j < 128; ++j) S[j] = 0.f;

    const float nw = norm_w[v];
    const size_t zoff = (size_t)nk * 768 + 512 + (size_t)(nv & 1) * 128 + v;

    for (int t = 0; t < L_SEQ; ++t) {
        const float* mrow = mixed + (size_t)t * CONV_DIM;
        qsh[v] = mrow[nk * DK_D + v];
        ksh[v] = mrow[KEY_DIM + nk * DK_D + v];
        const float vt = mrow[2 * KEY_DIM + nv * DV_D + v];
        const float gt = g[t * NV_H + nv];
        const float bt = beta[t * NV_H + nv];
        __syncthreads();                       // (1) k/q visible

        const float eg = __expf(gt);
        float kvdot = 0.f;
#pragma unroll
        for (int j = 0; j < 128; ++j) kvdot = fmaf(S[j], ksh[j], kvdot);
        const float delta = (vt - eg * kvdot) * bt;

        float o = 0.f;
#pragma unroll
        for (int j = 0; j < 128; ++j) {
            S[j] = fmaf(ksh[j], delta, S[j] * eg);
            o = fmaf(S[j], qsh[j], o);
        }

        float s2 = o * o;
#pragma unroll
        for (int off = 16; off > 0; off >>= 1) s2 += __shfl_down(s2, off, 32);
        if (lane == 0) red[wid] = s2;
        __syncthreads();                       // (2) partials visible
        const float mean = (red[0] + red[1] + red[2] + red[3]) * (1.0f / 128.0f);

        const float cn = o * rsqrtf(mean + EPS_RMS);
        const float z  = qkvz[(size_t)t * QKVZ_DIM + zoff];
        const float val = cn * nw * silu_f(z);
        cg[(size_t)t * VAL_DIM + nv * DV_D + v] = f32_to_bf16_rne(val);
    }
}

// ------------------------------ launcher -----------------------------------

extern "C" void kernel_launch(void* const* d_in, const int* in_sizes, int n_in,
                              void* d_out, int out_size, void* d_ws, size_t ws_size,
                              hipStream_t stream) {
    const float* hs      = (const float*)d_in[0];  // [L,H]
    const float* w_qkvz  = (const float*)d_in[1];  // [12288,H]
    const float* w_ba    = (const float*)d_in[2];  // [64,H]
    const float* conv_w  = (const float*)d_in[3];  // [CONV_DIM,4]
    const float* dt_bias = (const float*)d_in[4];  // [32]
    const float* a_log   = (const float*)d_in[5];  // [32]
    const float* norm_w  = (const float*)d_in[6];  // [128]
    const float* w_out   = (const float*)d_in[7];  // [H,VAL_DIM]
    float* out = (float*)d_out;                    // [L,H] f32

    char* ws = (char*)d_ws;
    size_t off = 0;
    auto take = [&](size_t bytes) {
        size_t o = off;
        off += (bytes + 255) & ~(size_t)255;
        return o;
    };
    unsigned short* hsb   = (unsigned short*)(ws + take((size_t)L_SEQ * H_DIM * 2));
    unsigned short* wqb   = (unsigned short*)(ws + take((size_t)QKVZ_DIM * H_DIM * 2));
    unsigned short* wob   = (unsigned short*)(ws + take((size_t)H_DIM * VAL_DIM * 2));
    float*          qkvz  = (float*)(ws + take((size_t)L_SEQ * QKVZ_DIM * 4));
    float*          ba    = (float*)(ws + take((size_t)L_SEQ * BA_DIM * 4));
    float*          mixed = (float*)(ws + take((size_t)L_SEQ * CONV_DIM * 4));
    float*          gbuf  = (float*)(ws + take((size_t)L_SEQ * NV_H * 4));
    float*          bbuf  = (float*)(ws + take((size_t)L_SEQ * NV_H * 4));
    unsigned short* cgb   = (unsigned short*)(ws + take((size_t)L_SEQ * VAL_DIM * 2));
    (void)ws_size; (void)in_sizes; (void)n_in; (void)out_size;

    // 1) f32 -> bf16 conversions (vectorized x4)
    {
        int n1 = (L_SEQ * H_DIM) / 4;
        cvt_f32_bf16<<<(n1 + 255) / 256, 256, 0, stream>>>(hs, hsb, n1);
        int n2 = (QKVZ_DIM * H_DIM) / 4;
        cvt_f32_bf16<<<(n2 + 255) / 256, 256, 0, stream>>>(w_qkvz, wqb, n2);
        int n3 = (H_DIM * VAL_DIM) / 4;
        cvt_f32_bf16<<<(n3 + 255) / 256, 256, 0, stream>>>(w_out, wob, n3);
    }

    // 2) qkvz = hs @ w_qkvz^T (WMMA bf16, M=4096 N=12288 K=2048)
    {
        dim3 grid(QKVZ_DIM / TILE_N, L_SEQ / TILE_M);
        gemm_bf16_wmma<<<grid, 256, 0, stream>>>(hsb, wqb, qkvz,
                                                 L_SEQ, QKVZ_DIM, H_DIM);
    }

    // 3) ba = hs @ w_ba^T (small)
    ba_gemm<<<L_SEQ, 64, 0, stream>>>(hs, w_ba, ba);

    // 4) causal conv + SiLU + L2-norm
    {
        dim3 grid(CONV_DIM / 128, L_SEQ);
        conv_silu_norm<<<grid, 128, 0, stream>>>(qkvz, conv_w, mixed);
    }

    // 5) per-step decay g and gate beta
    gb_kernel<<<L_SEQ, 32, 0, stream>>>(ba, dt_bias, a_log, gbuf, bbuf);

    // 6) recurrent delta-rule scan + fused RMSNorm/gating -> bf16
    scan_kernel<<<NV_H, 128, 0, stream>>>(mixed, qkvz, gbuf, bbuf, norm_w, cgb);

    // 7) out = cg @ w_out^T (WMMA bf16, M=4096 N=2048 K=4096)
    {
        dim3 grid(H_DIM / TILE_N, L_SEQ / TILE_M);
        gemm_bf16_wmma<<<grid, 256, 0, stream>>>(cgb, wob, out,
                                                 L_SEQ, H_DIM, VAL_DIM);
    }
}
